// TGN_50251117363834
// MI455X (gfx1250) — matrix operand order repaired and checked by reading, not compiled, not run on hardware
//
#include <hip/hip_runtime.h>
#include <hip/hip_bf16.h>
#include <math.h>

// ---------------------------------------------------------------------------
// MI455X (gfx1250) TGN forward.
//
// Roofline: ~92 GFLOP vs ~0.4 GB gathered reads (~17us @ 23.3 TB/s).  All
// GEMMs are pushed through v_wmma_f32_16x16x32_f16 (f16 in, f32 accum) so the
// kernel stays memory-bound.  The K/V projection + attention are fused per
// destination node to avoid a 126 MB Kf/Vf round-trip through HBM.
// ---------------------------------------------------------------------------

typedef _Float16 half_t;
typedef __attribute__((ext_vector_type(16))) _Float16 v16h;
typedef __attribute__((ext_vector_type(8)))  _Float16 v8h;
typedef __attribute__((ext_vector_type(8)))  float    v8f;

#define BB    4096
#define NDD   (3 * BB)          // 12288
#define KK    10
#define NTOT  (NDD * (1 + KK))  // 135168
#define DD    128

// ---------------------------------------------------------------------------
// WMMA helpers.  D = A(16x32 f16) x B(32x16 f16) + C(16x16 f32), wave32.
//
// A fragment (contiguous k order, not the ISA table order): lane l -> row = l&15, hi = l>>4.
//   v16h element j -> K = 16*hi + j  => two contiguous 16B loads
//   from a row-major tile:  [16*hi .. 16*hi+7] and [16*hi+8 .. 16*hi+15].
// B fragment (32x16): lane l -> col N = l&15, hi = l>>4; element j -> K =
//   16*hi + j.  With B[k][n] = W[n][k] (weights stored [out][in] row-major),
//   this is one contiguous 32B load per lane.
// C/D: lane l, vgpr v -> row M = v + 8*(l>>4), col N = l&15.
// ---------------------------------------------------------------------------

__device__ __forceinline__ v8f wmma16(v16h a, v16h b, v8f c) {
  return __builtin_amdgcn_wmma_f32_16x16x32_f16(
      /*neg_a=*/false, a, /*neg_b=*/false, b,
      /*c_mod=*/(short)0, c, /*reuse_a=*/false, /*reuse_b=*/false);
}

__device__ __forceinline__ v16h load_a_frag(const half_t* tile, int lda) {
  const int lane = threadIdx.x & 31;
  const int row  = lane & 15;
  const int hi   = lane >> 4;
  const half_t* r = tile + row * lda;
  v8h lo = *(const v8h*)(r + 16 * hi);
  v8h hv = *(const v8h*)(r + 16 * hi + 8);
  v16h out;
#pragma unroll
  for (int i = 0; i < 8; ++i) { out[i] = lo[i]; out[i + 8] = hv[i]; }
  return out;
}

__device__ __forceinline__ v16h load_b_frag(const half_t* W, int n0, int ldb, int k0) {
  const int lane = threadIdx.x & 31;
  const int col  = lane & 15;
  const int hi   = lane >> 4;
  return *(const v16h*)(W + (size_t)(n0 + col) * ldb + k0 + 16 * hi);
}

__device__ __forceinline__ float sigmoidf_(float x) { return 1.f / (1.f + expf(-x)); }

// ---------------------------------------------------------------------------
// Weight conversion kernels
// ---------------------------------------------------------------------------
__global__ void cvt_f32_to_f16(const float* __restrict__ src,
                               half_t* __restrict__ dst, int n) {
  int i = blockIdx.x * blockDim.x + threadIdx.x;
  if (i < n) dst[i] = (half_t)src[i];
}

// wq is (128, 256); only the first 128 input columns multiply dst_h.
__global__ void cvt_wq_kernel(const float* __restrict__ wq, half_t* __restrict__ dst) {
  int i = blockIdx.x * blockDim.x + threadIdx.x;
  if (i < 128 * 128) {
    int o = i >> 7, j = i & 127;
    dst[i] = (half_t)wq[o * 256 + j];
  }
}

// qconst[o] = bq[o] + sum_j wq[o][128+j] * cos(time_b[j])   (zero_tenc fold)
__global__ void qconst_kernel(const float* __restrict__ wq, const float* __restrict__ bq,
                              const float* __restrict__ time_b, float* __restrict__ qc) {
  int o = threadIdx.x;
  float s = bq[o];
  for (int j = 0; j < 128; ++j) s += wq[o * 256 + 128 + j] * cosf(time_b[j]);
  qc[o] = s;
}

// ---------------------------------------------------------------------------
// Kernel 1: GRU memory update + h = nfeat + mem.  16 nodes / block, 4 waves.
// gi = X[16x512] @ Wih.T  (16 WMMA k-steps per 16-col tile, 24 tiles)
// gh = Hm[16x128] @ Whh.T (4 k-steps per tile)
// LDS: Gs = gi+gh for r/z blocks, gi only for n block; Ghn = gh of n block.
// ---------------------------------------------------------------------------
__global__ void __launch_bounds__(128) gru_kernel(
    const int* __restrict__ nodes, const float* __restrict__ mail_time,
    const float* __restrict__ mem_time, const float* __restrict__ mail_data,
    const float* __restrict__ mem_data, const float* __restrict__ nfeat,
    const float* __restrict__ time_w, const float* __restrict__ time_b,
    const half_t* __restrict__ wih16, const half_t* __restrict__ whh16,
    const float* __restrict__ b_ih, const float* __restrict__ b_hh,
    half_t* __restrict__ Hh) {
  __shared__ __align__(32) half_t Xs[16][512];   // 16 KB: [mail(384) | tenc(128)]
  __shared__ __align__(32) half_t Hm[16][128];   // 4 KB
  __shared__ float Hf[16][128];                  // 8 KB
  __shared__ float Gs[16][384];                  // 24 KB
  __shared__ float Ghn[16][128];                 // 8 KB
  __shared__ int   idxs[16];
  __shared__ float dts[16];

  const int t  = threadIdx.x;
  const int r0 = blockIdx.x * 16;

  if (t < 16) {
    int idx = nodes[r0 + t];
    idxs[t] = idx;
    dts[t]  = mail_time[idx] - mem_time[idx];
  }
  __syncthreads();

  for (int i = t; i < 16 * 384; i += 128) {
    int r = i / 384, c = i - r * 384;
    Xs[r][c] = (half_t)mail_data[(size_t)idxs[r] * 384 + c];
  }
  for (int i = t; i < 16 * 128; i += 128) {
    int r = i >> 7, j = i & 127;
    Xs[r][384 + j] = (half_t)cosf(dts[r] * time_w[j] + time_b[j]);
    float m = mem_data[(size_t)idxs[r] * 128 + j];
    Hf[r][j] = m;
    Hm[r][j] = (half_t)m;
  }
  __syncthreads();

  const int wave = t >> 5;
  const int lane = t & 31, col = lane & 15, hi = lane >> 4;
#pragma unroll 1
  for (int i = 0; i < 6; ++i) {
    const int ct = wave * 6 + i, n0 = ct * 16;
    v8f gi = {};
#pragma unroll
    for (int ks = 0; ks < 16; ++ks)
      gi = wmma16(load_a_frag(&Xs[0][0] + ks * 32, 512),
                  load_b_frag(wih16, n0, 512, ks * 32), gi);
    v8f gh = {};
#pragma unroll
    for (int ks = 0; ks < 4; ++ks)
      gh = wmma16(load_a_frag(&Hm[0][0] + ks * 32, 128),
                  load_b_frag(whh16, n0, 128, ks * 32), gh);
    if (n0 < 256) {
#pragma unroll
      for (int v = 0; v < 8; ++v) Gs[v + 8 * hi][n0 + col] = gi[v] + gh[v];
    } else {
#pragma unroll
      for (int v = 0; v < 8; ++v) {
        Gs[v + 8 * hi][n0 + col]        = gi[v];
        Ghn[v + 8 * hi][n0 - 256 + col] = gh[v];
      }
    }
  }
  __syncthreads();

  for (int i = t; i < 16 * 128; i += 128) {
    int r = i >> 7, d = i & 127;
    float rg = sigmoidf_(Gs[r][d]       + b_ih[d]       + b_hh[d]);
    float z  = sigmoidf_(Gs[r][128 + d] + b_ih[128 + d] + b_hh[128 + d]);
    float nn = tanhf((Gs[r][256 + d] + b_ih[256 + d]) +
                     rg * (Ghn[r][d] + b_hh[256 + d]));
    float mem = (1.f - z) * nn + z * Hf[r][d];
    float h   = nfeat[(size_t)idxs[r] * 128 + d] + mem;
    Hh[(size_t)(r0 + r) * 128 + d] = (half_t)h;
  }
}

// ---------------------------------------------------------------------------
// Kernel 2: Q = dst_h @ wq[:, :128].T + qconst.  A loaded straight from
// global f16 h (row-major, pitch 128 halves = 256 B aligned rows).
// ---------------------------------------------------------------------------
__global__ void __launch_bounds__(128) q_kernel(
    const half_t* __restrict__ Hh, const half_t* __restrict__ wq16,
    const float* __restrict__ qconst, float* __restrict__ Qf) {
  const int t = threadIdx.x, wave = t >> 5;
  const int lane = t & 31, col = lane & 15, hi = lane >> 4;
  const size_t r0 = (size_t)blockIdx.x * 16;
  const half_t* A = Hh + r0 * 128;
#pragma unroll 1
  for (int i = 0; i < 2; ++i) {
    const int ct = wave * 2 + i, n0 = ct * 16;
    v8f acc = {};
#pragma unroll
    for (int ks = 0; ks < 4; ++ks)
      acc = wmma16(load_a_frag(A + ks * 32, 128),
                   load_b_frag(wq16, n0, 128, ks * 32), acc);
#pragma unroll
    for (int v = 0; v < 8; ++v) {
      int n = n0 + col;
      Qf[(r0 + v + 8 * hi) * 128 + n] = acc[v] + qconst[n];
    }
  }
}

// ---------------------------------------------------------------------------
// Kernel 3: fused K/V projection + 2-head softmax attention + aggregation.
// One block (4 waves) per destination node; 10 real neighbor rows padded to
// a 16-row WMMA tile (pad rows zeroed; WMMA rows are independent).
// ---------------------------------------------------------------------------
__global__ void __launch_bounds__(128) attn_kernel(
    const half_t* __restrict__ Hh, const int* __restrict__ eids,
    const float* __restrict__ dt_nbr, const float* __restrict__ efeat,
    const float* __restrict__ time_w, const float* __restrict__ time_b,
    const half_t* __restrict__ wk16, const half_t* __restrict__ wv16,
    const float* __restrict__ bk, const float* __restrict__ bv,
    const float* __restrict__ Qf, float* __restrict__ Agg) {
  __shared__ __align__(32) half_t Xs[16][384];  // [src_h | efeat | tenc]
  __shared__ float Kf[16][128];
  __shared__ float Vf[16][128];
  __shared__ float Qrow[128];
  __shared__ float att[2][16];
  __shared__ int   eid_s[16];
  __shared__ float dt_s[16];

  const int t = threadIdx.x;
  const int n = blockIdx.x;

  if (t < KK) {
    eid_s[t] = eids[n * KK + t];
    dt_s[t]  = dt_nbr[n * KK + t];
  }
  Qrow[t] = Qf[(size_t)n * 128 + t];
  __syncthreads();

  for (int i = t; i < 16 * 384; i += 128) {
    int r = i / 384, c = i - r * 384;
    half_t v = (half_t)0.f;
    if (r < KK) {
      if (c < 128)
        v = Hh[(size_t)(NDD + n * KK + r) * 128 + c];
      else if (c < 256)
        v = (half_t)efeat[(size_t)eid_s[r] * 128 + (c - 128)];
      else {
        int j = c - 256;
        v = (half_t)cosf(dt_s[r] * time_w[j] + time_b[j]);
      }
    }
    Xs[r][c] = v;
  }
  __syncthreads();

  const int wave = t >> 5;
  const int lane = t & 31, col = lane & 15, hi = lane >> 4;
#pragma unroll 1
  for (int i = 0; i < 2; ++i) {
    const int ct = wave * 2 + i, n0 = ct * 16;
    v8f ak = {};
#pragma unroll
    for (int ks = 0; ks < 12; ++ks)
      ak = wmma16(load_a_frag(&Xs[0][0] + ks * 32, 384),
                  load_b_frag(wk16, n0, 384, ks * 32), ak);
#pragma unroll
    for (int v = 0; v < 8; ++v) Kf[v + 8 * hi][n0 + col] = ak[v] + bk[n0 + col];
    v8f av = {};
#pragma unroll
    for (int ks = 0; ks < 12; ++ks)
      av = wmma16(load_a_frag(&Xs[0][0] + ks * 32, 384),
                  load_b_frag(wv16, n0, 384, ks * 32), av);
#pragma unroll
    for (int v = 0; v < 8; ++v) Vf[v + 8 * hi][n0 + col] = av[v] + bv[n0 + col];
  }
  __syncthreads();

  if (t < 2 * KK) {  // 20 logits: (k, head)
    int k = t >> 1, h = t & 1;
    float s = 0.f;
    for (int d = 0; d < 64; ++d) s += Qrow[h * 64 + d] * Kf[k][h * 64 + d];
    att[h][k] = s;
  }
  __syncthreads();
  if (t < 2) {  // per-head softmax over K
    float m = -1e30f;
    for (int k = 0; k < KK; ++k) m = fmaxf(m, att[t][k]);
    float e[KK], sum = 0.f;
    for (int k = 0; k < KK; ++k) { e[k] = expf(att[t][k] - m); sum += e[k]; }
    float inv = 1.f / sum;
    for (int k = 0; k < KK; ++k) att[t][k] = e[k] * inv;
  }
  __syncthreads();
  {
    int h = t >> 6;
    float s = 0.f;
    for (int k = 0; k < KK; ++k) s += att[h][k] * Vf[k][t];
    Agg[(size_t)n * 128 + t] = s;
  }
}

// ---------------------------------------------------------------------------
// Kernel 4: out = relu([dst_h | agg] @ wo.T + bo); layernorm -> emb (f16).
// ---------------------------------------------------------------------------
__global__ void __launch_bounds__(128) out_kernel(
    const half_t* __restrict__ Hh, const float* __restrict__ Agg,
    const half_t* __restrict__ wo16, const float* __restrict__ bo,
    const float* __restrict__ ln_g, const float* __restrict__ ln_b,
    half_t* __restrict__ Emb16) {
  __shared__ __align__(32) half_t Xs[16][256];
  __shared__ float Out[16][128];
  __shared__ float mu_s[16], rs_s[16];

  const int t = threadIdx.x;
  const size_t r0 = (size_t)blockIdx.x * 16;

  for (int i = t; i < 16 * 256; i += 128) {
    int r = i >> 8, c = i & 255;
    Xs[r][c] = (c < 128) ? Hh[(r0 + r) * 128 + c]
                         : (half_t)Agg[(r0 + r) * 128 + (c - 128)];
  }
  __syncthreads();

  const int wave = t >> 5;
  const int lane = t & 31, col = lane & 15, hi = lane >> 4;
#pragma unroll 1
  for (int i = 0; i < 2; ++i) {
    const int ct = wave * 2 + i, n0 = ct * 16;
    v8f acc = {};
#pragma unroll
    for (int ks = 0; ks < 8; ++ks)
      acc = wmma16(load_a_frag(&Xs[0][0] + ks * 32, 256),
                   load_b_frag(wo16, n0, 256, ks * 32), acc);
#pragma unroll
    for (int v = 0; v < 8; ++v)
      Out[v + 8 * hi][n0 + col] = fmaxf(acc[v] + bo[n0 + col], 0.f);
  }
  __syncthreads();

  if (t < 16) {
    float s = 0.f, s2 = 0.f;
    for (int d = 0; d < 128; ++d) { float x = Out[t][d]; s += x; s2 += x * x; }
    float mu  = s * (1.f / 128.f);
    float var = s2 * (1.f / 128.f) - mu * mu;
    mu_s[t] = mu;
    rs_s[t] = rsqrtf(var + 1e-5f);
  }
  __syncthreads();

  for (int i = t; i < 16 * 128; i += 128) {
    int r = i >> 7, d = i & 127;
    float e = ln_g[d] * (Out[r][d] - mu_s[r]) * rs_s[r] + ln_b[d];
    Emb16[(r0 + r) * 128 + d] = (half_t)e;
  }
}

// ---------------------------------------------------------------------------
// Kernel 5: edge-predictor projections (src/dst/neg) via WMMA.
// 24 (matrix, col-tile) pairs spread over 4 waves; `which` is wave-uniform.
// ---------------------------------------------------------------------------
__global__ void __launch_bounds__(128) edgeproj_kernel(
    const half_t* __restrict__ Emb16, const half_t* __restrict__ epsw16,
    const half_t* __restrict__ epdw16, const float* __restrict__ ep_sb,
    const float* __restrict__ ep_db, float* __restrict__ SP,
    float* __restrict__ DP, float* __restrict__ NP) {
  const int t = threadIdx.x, wave = t >> 5;
  const int lane = t & 31, col = lane & 15, hi = lane >> 4;
  const size_t r0 = (size_t)blockIdx.x * 16;
#pragma unroll 1
  for (int i = 0; i < 6; ++i) {
    const int tt = wave * 6 + i;
    const int which = tt >> 3, ct = tt & 7, n0 = ct * 16;
    const half_t* A    = Emb16 + ((size_t)which * BB + r0) * 128;
    const half_t* W    = (which == 0) ? epsw16 : epdw16;
    const float*  bias = (which == 0) ? ep_sb : ep_db;
    float* Dst = (which == 0) ? SP : ((which == 1) ? DP : NP);
    v8f acc = {};
#pragma unroll
    for (int ks = 0; ks < 4; ++ks)
      acc = wmma16(load_a_frag(A + ks * 32, 128),
                   load_b_frag(W, n0, 128, ks * 32), acc);
#pragma unroll
    for (int v = 0; v < 8; ++v) {
      int nn = n0 + col;
      Dst[(r0 + v + 8 * hi) * 128 + nn] = acc[v] + bias[nn];
    }
  }
}

// ---------------------------------------------------------------------------
// Kernel 6: pos/neg = relu(sproj + {d,n}proj) . ep_ow + ep_ob -> d_out[B][2]
// ---------------------------------------------------------------------------
__global__ void __launch_bounds__(128) final_kernel(
    const float* __restrict__ SP, const float* __restrict__ DP,
    const float* __restrict__ NP, const float* __restrict__ ep_ow,
    const float* __restrict__ ep_ob, float* __restrict__ out) {
  __shared__ float red[128];
  const int t = threadIdx.x, n = blockIdx.x;
  const float sp = SP[(size_t)n * 128 + t];
  const float ow = ep_ow[t];

  red[t] = fmaxf(sp + DP[(size_t)n * 128 + t], 0.f) * ow;
  __syncthreads();
  for (int s = 64; s >= 1; s >>= 1) {
    if (t < s) red[t] += red[t + s];
    __syncthreads();
  }
  const float pos = red[0];
  __syncthreads();

  red[t] = fmaxf(sp + NP[(size_t)n * 128 + t], 0.f) * ow;
  __syncthreads();
  for (int s = 64; s >= 1; s >>= 1) {
    if (t < s) red[t] += red[t + s];
    __syncthreads();
  }
  if (t == 0) {
    out[n * 2 + 0] = pos + ep_ob[0];
    out[n * 2 + 1] = red[0] + ep_ob[0];
  }
}

// ---------------------------------------------------------------------------
extern "C" void kernel_launch(void* const* d_in, const int* in_sizes, int n_in,
                              void* d_out, int out_size, void* d_ws, size_t ws_size,
                              hipStream_t stream) {
  (void)in_sizes; (void)n_in; (void)out_size; (void)ws_size;

  const int*   nodes     = (const int*)d_in[0];
  const int*   eids      = (const int*)d_in[1];
  const float* dt_nbr    = (const float*)d_in[2];
  const float* nfeat     = (const float*)d_in[3];
  const float* mem_data  = (const float*)d_in[4];
  const float* mem_time  = (const float*)d_in[5];
  const float* mail_data = (const float*)d_in[6];
  const float* mail_time = (const float*)d_in[7];
  const float* efeat     = (const float*)d_in[8];
  const float* time_w    = (const float*)d_in[9];
  const float* time_b    = (const float*)d_in[10];
  const float* gru_w_ih  = (const float*)d_in[11];
  const float* gru_w_hh  = (const float*)d_in[12];
  const float* gru_b_ih  = (const float*)d_in[13];
  const float* gru_b_hh  = (const float*)d_in[14];
  const float* wq        = (const float*)d_in[15];
  const float* wk        = (const float*)d_in[17];
  const float* bk        = (const float*)d_in[18];
  const float* wv        = (const float*)d_in[19];
  const float* bv        = (const float*)d_in[20];
  const float* wo        = (const float*)d_in[21];
  const float* bo        = (const float*)d_in[22];
  const float* ln_g      = (const float*)d_in[23];
  const float* ln_b      = (const float*)d_in[24];
  const float* ep_sw     = (const float*)d_in[25];
  const float* ep_sb     = (const float*)d_in[26];
  const float* ep_dw     = (const float*)d_in[27];
  const float* ep_db     = (const float*)d_in[28];
  const float* ep_ow     = (const float*)d_in[29];
  const float* ep_ob     = (const float*)d_in[30];
  const float* bq        = (const float*)d_in[16];
  float* outp = (float*)d_out;

  char* ws = (char*)d_ws;
  size_t off = 0;
  auto take = [&](size_t bytes) -> char* {
    char* p = ws + off;
    off = (off + bytes + 255) & ~(size_t)255;
    return p;
  };

  half_t* wih16  = (half_t*)take((size_t)384 * 512 * 2);
  half_t* whh16  = (half_t*)take((size_t)384 * 128 * 2);
  half_t* wq16   = (half_t*)take((size_t)128 * 128 * 2);
  half_t* wk16   = (half_t*)take((size_t)128 * 384 * 2);
  half_t* wv16   = (half_t*)take((size_t)128 * 384 * 2);
  half_t* wo16   = (half_t*)take((size_t)128 * 256 * 2);
  half_t* epsw16 = (half_t*)take((size_t)128 * 128 * 2);
  half_t* epdw16 = (half_t*)take((size_t)128 * 128 * 2);
  float*  qconst = (float*)take(128 * 4);
  half_t* Hh     = (half_t*)take((size_t)NTOT * 128 * 2);   // 34.6 MB
  float*  Qf     = (float*)take((size_t)NDD * 128 * 4);     // 6.3 MB
  float*  Agg    = (float*)take((size_t)NDD * 128 * 4);     // 6.3 MB
  half_t* Emb16  = (half_t*)take((size_t)NDD * 128 * 2);    // 3.1 MB
  float*  SP     = (float*)take((size_t)BB * 128 * 4);
  float*  DP     = (float*)take((size_t)BB * 128 * 4);
  float*  NP     = (float*)take((size_t)BB * 128 * 4);

  // --- weight precision conversion (tiny) ---
  cvt_f32_to_f16<<<(384 * 512 + 255) / 256, 256, 0, stream>>>(gru_w_ih, wih16, 384 * 512);
  cvt_f32_to_f16<<<(384 * 128 + 255) / 256, 256, 0, stream>>>(gru_w_hh, whh16, 384 * 128);
  cvt_f32_to_f16<<<(128 * 384 + 255) / 256, 256, 0, stream>>>(wk, wk16, 128 * 384);
  cvt_f32_to_f16<<<(128 * 384 + 255) / 256, 256, 0, stream>>>(wv, wv16, 128 * 384);
  cvt_f32_to_f16<<<(128 * 256 + 255) / 256, 256, 0, stream>>>(wo, wo16, 128 * 256);
  cvt_f32_to_f16<<<(128 * 128 + 255) / 256, 256, 0, stream>>>(ep_sw, epsw16, 128 * 128);
  cvt_f32_to_f16<<<(128 * 128 + 255) / 256, 256, 0, stream>>>(ep_dw, epdw16, 128 * 128);
  cvt_wq_kernel<<<(128 * 128 + 255) / 256, 256, 0, stream>>>(wq, wq16);
  qconst_kernel<<<1, 128, 0, stream>>>(wq, bq, time_b, qconst);

  // --- main pipeline ---
  gru_kernel<<<NTOT / 16, 128, 0, stream>>>(nodes, mail_time, mem_time, mail_data,
                                            mem_data, nfeat, time_w, time_b,
                                            wih16, whh16, gru_b_ih, gru_b_hh, Hh);
  q_kernel<<<NDD / 16, 128, 0, stream>>>(Hh, wq16, qconst, Qf);
  attn_kernel<<<NDD, 128, 0, stream>>>(Hh, eids, dt_nbr, efeat, time_w, time_b,
                                       wk16, wv16, bk, bv, Qf, Agg);
  out_kernel<<<NDD / 16, 128, 0, stream>>>(Hh, Agg, wo16, bo, ln_g, ln_b, Emb16);
  edgeproj_kernel<<<BB / 16, 128, 0, stream>>>(Emb16, epsw16, epdw16, ep_sb, ep_db,
                                               SP, DP, NP);
  final_kernel<<<BB, 128, 0, stream>>>(SP, DP, NP, ep_ow, ep_ob, outp);
}
